// StrongGNNModel_16612933501368
// MI455X (gfx1250) — matrix operand order repaired and checked by reading, not compiled, not run on hardware
//
#include <hip/hip_runtime.h>
#include <hip/hip_bf16.h>

#define N_NODES   100000
#define N_EDGES   200000
#define N_GRAPHS  2000
#define HID       128
#define EEMB      64
#define NLAYERS   5
#define NODE_IN   32
#define EXP_IN    200
#define BN_SCALE  0.9999950000374996f   // 1/sqrt(1+1e-5)

typedef __attribute__((ext_vector_type(16))) __bf16 v16bf;
typedef __attribute__((ext_vector_type(8)))  float  v8f;

// ---------------------------------------------------------------- WMMA helpers
__device__ inline v8f wmma_bf16(v16bf a, v16bf b, v8f c) {
  return __builtin_amdgcn_wmma_f32_16x16x32_bf16(
      /*neg_a=*/false, a, /*neg_b=*/false, b,
      /*c_mod=*/(short)0, c, /*reuse_a=*/false, /*reuse_b=*/false);
}

// Swizzled LDS layouts: fragments are stored per-lane contiguous (16 bf16 =
// 32B per lane) so a fragment load is two ds_load_b128 instead of 16 u16 loads.
//
// B element (k, c) of a (K x 128) weight matrix:
//   tile = (k>>5)*8 + (c>>4) ; lane = ((k>>4)&1)*16 + (c&15) ; elem = k&15
__device__ inline int b_swz_idx(int k, int c) {
  int tile = ((k >> 5) << 3) + (c >> 4);
  int lane = (((k >> 4) & 1) << 4) + (c & 15);
  return ((tile << 5) + lane) * 16 + (k & 15);
}
// A element (m, k) of a (16 x K) activation tile:
//   tile = k>>5 ; lane = ((k>>4)&1)*16 + m ; elem = k&15
__device__ inline int a_swz_idx(int m, int k) {
  int kt = k >> 5, kl = k & 31;
  int g = (kl >> 4) & 1;
  int i = kl & 15;
  return ((kt << 5) + (g << 4) + m) * 16 + i;
}

__device__ inline v16bf frag_ld(const __bf16* base32B) {
  return *(const v16bf*)base32B;   // lowers to 2x ds_load_b128
}

// ordered-int mapping for float atomic max
__device__ inline unsigned f2ord(float f) {
  unsigned u = __float_as_uint(f);
  return (u & 0x80000000u) ? ~u : (u | 0x80000000u);
}
__device__ inline float ord2f(unsigned u) {
  unsigned v = (u & 0x80000000u) ? (u & 0x7FFFFFFFu) : ~u;
  return __uint_as_float(v);
}

// ---------------------------------------------------------------- utility
__global__ void k_zero(float* __restrict__ p, long n) {
  long i = (long)blockIdx.x * blockDim.x + threadIdx.x;
  long stride = (long)gridDim.x * blockDim.x;
  for (; i < n; i += stride) p[i] = 0.f;
}

// ---------------------------------------------------------------- edge table
// tab[l][t][d][h] = (te[t]+de[d]) @ lew[l] + leb[l]
__global__ void k_tab(const float* __restrict__ te, const float* __restrict__ de,
                      const float* __restrict__ lew, const float* __restrict__ leb,
                      float* __restrict__ tab) {
  int idx = blockIdx.x * blockDim.x + threadIdx.x;
  const int total = NLAYERS * 24 * HID;
  if (idx >= total) return;
  int hcol = idx & (HID - 1);
  int comb = idx >> 7;           // l*24 + c
  int l = comb / 24, c = comb % 24;
  int t = c >> 2, d = c & 3;
  float s = leb[l * HID + hcol];
  for (int k = 0; k < EEMB; ++k)
    s += (te[t * EEMB + k] + de[d * EEMB + k]) * lew[(l * EEMB + k) * HID + hcol];
  tab[idx] = s;
}

// cooperative: stage a (K x 128) f32 weight matrix into swizzled bf16 LDS
__device__ inline void stage_weights(__bf16* W, const float* __restrict__ w,
                                     int K, int tid, int nthreads) {
  for (int i = tid * 8; i < K * HID; i += nthreads * 8) {
    int k = i >> 7, c0 = i & 127;
    float4 a = *(const float4*)(w + i);
    float4 b = *(const float4*)(w + i + 4);
    W[b_swz_idx(k, c0 + 0)] = (__bf16)a.x;
    W[b_swz_idx(k, c0 + 1)] = (__bf16)a.y;
    W[b_swz_idx(k, c0 + 2)] = (__bf16)a.z;
    W[b_swz_idx(k, c0 + 3)] = (__bf16)a.w;
    W[b_swz_idx(k, c0 + 4)] = (__bf16)b.x;
    W[b_swz_idx(k, c0 + 5)] = (__bf16)b.y;
    W[b_swz_idx(k, c0 + 6)] = (__bf16)b.z;
    W[b_swz_idx(k, c0 + 7)] = (__bf16)b.w;
  }
}

// ---------------------------------------------------------------- node proj (WMMA)
// h = relu(x @ nw1 + nb1) @ nw2 + nb2 ; 4 waves/block, 16 rows/wave
__global__ void k_nodeproj(const float* __restrict__ x,
                           const float* __restrict__ nw1, const float* __restrict__ nb1,
                           const float* __restrict__ nw2, const float* __restrict__ nb2,
                           float* __restrict__ h, int nrows) {
  extern __shared__ char smem[];
  __bf16* W  = (__bf16*)smem;                 // swizzled weights, 32KB max
  __bf16* AT = (__bf16*)(smem + 32768);       // 4 waves * 4KB swizzled A tiles
  int tid = threadIdx.x, wave = tid >> 5, lane = tid & 31;
  int g = (lane >> 4) & 1, n = lane & 15;

  stage_weights(W, nw1, NODE_IN, tid, blockDim.x);
  __syncthreads();

  long row0 = ((long)blockIdx.x * 4 + wave) * 16;
  __bf16* A = AT + wave * (4 * 32 * 16);

  // stage x tile 16x32 into swizzled A (kt = 0)
  for (int i = lane; i < 16 * NODE_IN; i += 32) {
    int r = i >> 5, c = i & 31;
    long row = row0 + r;
    A[a_swz_idx(r, c)] = (__bf16)((row < nrows) ? x[row * NODE_IN + c] : 0.f);
  }
  v16bf Af = frag_ld(A + lane * 16);

  v8f acc[8];
#pragma unroll
  for (int ct = 0; ct < 8; ++ct) {
    float b = nb1[ct * 16 + n];
#pragma unroll
    for (int r = 0; r < 8; ++r) acc[ct][r] = b;
    v16bf Bf = frag_ld(W + (ct * 32 + lane) * 16);   // tile = 0*8+ct
    acc[ct] = wmma_bf16(Af, Bf, acc[ct]);
  }
  // relu -> tmp, written directly in swizzled A layout
#pragma unroll
  for (int ct = 0; ct < 8; ++ct)
#pragma unroll
    for (int r = 0; r < 8; ++r)
      A[a_swz_idx(r + 8 * g, ct * 16 + n)] = (__bf16)fmaxf(acc[ct][r], 0.f);

  __syncthreads();
  stage_weights(W, nw2, HID, tid, blockDim.x);
  __syncthreads();

  v16bf A2[4];
#pragma unroll
  for (int kt = 0; kt < 4; ++kt) A2[kt] = frag_ld(A + (kt * 32 + lane) * 16);
#pragma unroll
  for (int ct = 0; ct < 8; ++ct) {
    float b = nb2[ct * 16 + n];
#pragma unroll
    for (int r = 0; r < 8; ++r) acc[ct][r] = b;
#pragma unroll
    for (int kt = 0; kt < 4; ++kt) {
      v16bf Bf = frag_ld(W + ((kt * 8 + ct) * 32 + lane) * 16);
      acc[ct] = wmma_bf16(A2[kt], Bf, acc[ct]);
    }
  }
#pragma unroll
  for (int ct = 0; ct < 8; ++ct)
#pragma unroll
    for (int r = 0; r < 8; ++r) {
      long row = row0 + r + 8 * g;
      if (row < nrows) h[row * HID + ct * 16 + n] = acc[ct][r];
    }
}

// ---------------------------------------------------------------- edge phase
// one wave per edge: m = relu(h[src] + tab[combo]); atomic add into aggr[dst]
__global__ void k_edge(const float* __restrict__ h, const int* __restrict__ ei,
                       const int* __restrict__ ea, const float* __restrict__ tabl,
                       float* __restrict__ aggr) {
  int tid = blockIdx.x * blockDim.x + threadIdx.x;
  int e = tid >> 5, lane = tid & 31;
  if (e >= N_EDGES) return;
  int src = ei[e];
  int dst = ei[N_EDGES + e];
  int bt = ea[2 * e];     bt = bt < 0 ? 0 : (bt > 5 ? 5 : bt);
  int bd = ea[2 * e + 1]; bd = bd < 0 ? 0 : (bd > 3 ? 3 : bd);
  const float* t = tabl + (bt * 4 + bd) * HID;
  const float4 hv = *(const float4*)(h + (size_t)src * HID + lane * 4);
  const float4 tv = *(const float4*)(t + lane * 4);
  float* ad = aggr + (size_t)dst * HID + lane * 4;
  atomicAdd(ad + 0, fmaxf(hv.x + tv.x, 0.f));
  atomicAdd(ad + 1, fmaxf(hv.y + tv.y, 0.f));
  atomicAdd(ad + 2, fmaxf(hv.z + tv.z, 0.f));
  atomicAdd(ad + 3, fmaxf(hv.w + tv.w, 0.f));
}

// ---------------------------------------------------------------- node MLP (WMMA)
// z = h + aggr ; z = relu(z@mw1+mb1)@mw2+mb2 ; z = z*gamma*s+beta ; h = relu(z)+h_in
__global__ void k_mlp(float* __restrict__ h, const float* __restrict__ aggr,
                      const float* __restrict__ mw1, const float* __restrict__ mb1,
                      const float* __restrict__ mw2, const float* __restrict__ mb2,
                      const float* __restrict__ gamma, const float* __restrict__ beta,
                      int nrows) {
  extern __shared__ char smem[];
  __bf16* W  = (__bf16*)smem;                 // swizzled 128x128 bf16 (32KB), phased
  __bf16* AT = (__bf16*)(smem + 32768);       // 4 waves * 4KB swizzled A tiles
  int tid = threadIdx.x, wave = tid >> 5, lane = tid & 31;
  int g = (lane >> 4) & 1, n = lane & 15;

  stage_weights(W, mw1, HID, tid, blockDim.x);
  __syncthreads();

  long row0 = ((long)blockIdx.x * 4 + wave) * 16;
  __bf16* A = AT + wave * (4 * 32 * 16);

  // stage z = h + aggr (bf16) in swizzled A layout
  for (int i = lane; i < 16 * HID; i += 32) {
    int r = i >> 7, c = i & 127;
    long row = row0 + r;
    float v = 0.f;
    if (row < nrows) v = h[row * HID + c] + aggr[row * HID + c];
    A[a_swz_idx(r, c)] = (__bf16)v;
  }

  v16bf Af[4];
#pragma unroll
  for (int kt = 0; kt < 4; ++kt) Af[kt] = frag_ld(A + (kt * 32 + lane) * 16);

  v8f acc[8];
#pragma unroll
  for (int ct = 0; ct < 8; ++ct) {
    float b = mb1[ct * 16 + n];
#pragma unroll
    for (int r = 0; r < 8; ++r) acc[ct][r] = b;
#pragma unroll
    for (int kt = 0; kt < 4; ++kt) {
      v16bf Bf = frag_ld(W + ((kt * 8 + ct) * 32 + lane) * 16);
      acc[ct] = wmma_bf16(Af[kt], Bf, acc[ct]);
    }
  }
  // relu -> tmp, swizzled A layout
#pragma unroll
  for (int ct = 0; ct < 8; ++ct)
#pragma unroll
    for (int r = 0; r < 8; ++r)
      A[a_swz_idx(r + 8 * g, ct * 16 + n)] = (__bf16)fmaxf(acc[ct][r], 0.f);

  __syncthreads();
  stage_weights(W, mw2, HID, tid, blockDim.x);
  __syncthreads();

#pragma unroll
  for (int kt = 0; kt < 4; ++kt) Af[kt] = frag_ld(A + (kt * 32 + lane) * 16);
#pragma unroll
  for (int ct = 0; ct < 8; ++ct) {
    float b = mb2[ct * 16 + n];
#pragma unroll
    for (int r = 0; r < 8; ++r) acc[ct][r] = b;
#pragma unroll
    for (int kt = 0; kt < 4; ++kt) {
      v16bf Bf = frag_ld(W + ((kt * 8 + ct) * 32 + lane) * 16);
      acc[ct] = wmma_bf16(Af[kt], Bf, acc[ct]);
    }
  }
  // epilogue: BN affine + relu + residual, in-place on h
#pragma unroll
  for (int ct = 0; ct < 8; ++ct) {
    int col = ct * 16 + n;
    float ga = gamma[col] * BN_SCALE;
    float be = beta[col];
#pragma unroll
    for (int r = 0; r < 8; ++r) {
      long row = row0 + r + 8 * g;
      if (row < nrows) {
        float z  = acc[ct][r] * ga + be;
        float hn = fmaxf(z, 0.f) + h[row * HID + col];
        h[row * HID + col] = hn;
      }
    }
  }
}

// ---------------------------------------------------------------- pooling
__global__ void k_pool(const float* __restrict__ h, const int* __restrict__ batch,
                       float* __restrict__ psum, unsigned* __restrict__ maxb,
                       float* __restrict__ cnt) {
  int gid = blockIdx.x * blockDim.x + threadIdx.x;
  int node = gid >> 5, q = gid & 31;
  if (node >= N_NODES) return;
  int gg = batch[node];
  const float* hr = h + (size_t)node * HID + q * 4;
#pragma unroll
  for (int j = 0; j < 4; ++j) {
    float v = hr[j];
    atomicAdd(&psum[(size_t)gg * HID + q * 4 + j], v);
    atomicMax(&maxb[(size_t)gg * HID + q * 4 + j], f2ord(v));
  }
  if (q == 0) atomicAdd(&cnt[gg], 1.0f);
}

__global__ void k_poolfin(const float* __restrict__ psum, const unsigned* __restrict__ maxb,
                          const float* __restrict__ cnt, float* __restrict__ emb,
                          float* __restrict__ comb) {
  int idx = blockIdx.x * blockDim.x + threadIdx.x;
  if (idx >= N_GRAPHS * HID) return;
  int gg = idx >> 7, c = idx & (HID - 1);
  float mean = psum[idx] / cnt[gg];
  float mx = ord2f(maxb[idx]);
  emb[(size_t)gg * 256 + c]        = mean;
  emb[(size_t)gg * 256 + HID + c]  = mx;
  comb[(size_t)gg * 384 + c]       = mean;
  comb[(size_t)gg * 384 + HID + c] = mx;
}

// ---------------------------------------------------------------- experimental MLP
__global__ void k_exp(const float* __restrict__ expf,
                      const float* __restrict__ ew1, const float* __restrict__ eb1,
                      const float* __restrict__ ew2, const float* __restrict__ eb2,
                      float* __restrict__ comb) {
  __shared__ float xg[EXP_IN];
  __shared__ float t1[HID];
  int gg = blockIdx.x, tid = threadIdx.x;
  for (int i = tid; i < EXP_IN; i += blockDim.x) xg[i] = expf[(size_t)gg * EXP_IN + i];
  __syncthreads();
  float s = eb1[tid];
  for (int k = 0; k < EXP_IN; ++k) s += xg[k] * ew1[k * HID + tid];
  t1[tid] = fmaxf(s, 0.f);
  __syncthreads();
  float s2 = eb2[tid];
  for (int k = 0; k < HID; ++k) s2 += t1[k] * ew2[k * HID + tid];
  comb[(size_t)gg * 384 + 256 + tid] = fmaxf(s2, 0.f);
}

// ---------------------------------------------------------------- head
__global__ void k_head(const float* __restrict__ comb,
                       const float* __restrict__ hw1, const float* __restrict__ hb1,
                       const float* __restrict__ hw2, const float* __restrict__ hb2,
                       const float* __restrict__ hw3, const float* __restrict__ hb3,
                       float* __restrict__ outv) {
  __shared__ float cb[384];
  __shared__ float t1[256];
  __shared__ float t2[HID];
  int gg = blockIdx.x, tid = threadIdx.x;
  for (int i = tid; i < 384; i += blockDim.x) cb[i] = comb[(size_t)gg * 384 + i];
  __syncthreads();
  {
    float s = hb1[tid];
    for (int k = 0; k < 384; ++k) s += cb[k] * hw1[k * 256 + tid];
    t1[tid] = fmaxf(s, 0.f);
  }
  __syncthreads();
  if (tid < HID) {
    float s = hb2[tid];
    for (int k = 0; k < 256; ++k) s += t1[k] * hw2[k * HID + tid];
    t2[tid] = fmaxf(s, 0.f);
  }
  __syncthreads();
  if (tid == 0) {
    float s = hb3[0];
    for (int k = 0; k < HID; ++k) s += t2[k] * hw3[k];
    outv[gg] = s;
  }
}

// ---------------------------------------------------------------- launch
extern "C" void kernel_launch(void* const* d_in, const int* in_sizes, int n_in,
                              void* d_out, int out_size, void* d_ws, size_t ws_size,
                              hipStream_t stream) {
  (void)in_sizes; (void)n_in; (void)out_size; (void)ws_size;

  const float* x     = (const float*)d_in[0];
  const int*   ei    = (const int*)  d_in[1];
  const int*   ea    = (const int*)  d_in[2];
  const int*   batch = (const int*)  d_in[3];
  const float* expf  = (const float*)d_in[4];
  const float* nw1 = (const float*)d_in[5],  *nb1 = (const float*)d_in[6];
  const float* nw2 = (const float*)d_in[7],  *nb2 = (const float*)d_in[8];
  const float* te  = (const float*)d_in[9],  *de  = (const float*)d_in[10];
  const float* lew = (const float*)d_in[11], *leb = (const float*)d_in[12];
  const float* mw1 = (const float*)d_in[13], *mb1 = (const float*)d_in[14];
  const float* mw2 = (const float*)d_in[15], *mb2 = (const float*)d_in[16];
  const float* gam = (const float*)d_in[17], *bet = (const float*)d_in[18];
  const float* ew1 = (const float*)d_in[19], *eb1 = (const float*)d_in[20];
  const float* ew2 = (const float*)d_in[21], *eb2 = (const float*)d_in[22];
  const float* hw1 = (const float*)d_in[23], *hb1 = (const float*)d_in[24];
  const float* hw2 = (const float*)d_in[25], *hb2 = (const float*)d_in[26];
  const float* hw3 = (const float*)d_in[27], *hb3 = (const float*)d_in[28];

  // workspace carve (256B aligned): ~105 MB total
  char* ws = (char*)d_ws;
  size_t off = 0;
  auto carve = [&](size_t bytes) -> char* {
    char* p = ws + off;
    off = (off + bytes + 255) & ~(size_t)255;
    return p;
  };
  float*    h    = (float*)   carve((size_t)N_NODES * HID * 4);
  float*    aggr = (float*)   carve((size_t)N_NODES * HID * 4);
  float*    tab  = (float*)   carve((size_t)NLAYERS * 24 * HID * 4);
  float*    psum = (float*)   carve((size_t)N_GRAPHS * HID * 4);
  unsigned* maxb = (unsigned*)carve((size_t)N_GRAPHS * HID * 4);
  float*    cnt  = (float*)   carve((size_t)N_GRAPHS * 4);

  float* outv = (float*)d_out;
  float* emb  = outv + N_GRAPHS;                          // [G,256]
  float* comb = outv + N_GRAPHS + (size_t)N_GRAPHS * 256; // [G,384]

  const int SHM = 32768 + 4 * 4 * 32 * 16 * 2;  // 48KB: weights + 4 wave A-tiles
  const int nodeBlocks = (N_NODES + 63) / 64;

  k_tab<<<(NLAYERS * 24 * HID + 255) / 256, 256, 0, stream>>>(te, de, lew, leb, tab);
  k_nodeproj<<<nodeBlocks, 128, SHM, stream>>>(x, nw1, nb1, nw2, nb2, h, N_NODES);

  for (int l = 0; l < NLAYERS; ++l) {
    k_zero<<<2048, 256, 0, stream>>>(aggr, (long)N_NODES * HID);
    k_edge<<<(N_EDGES + 7) / 8, 256, 0, stream>>>(h, ei, ea, tab + (size_t)l * 24 * HID, aggr);
    k_mlp<<<nodeBlocks, 128, SHM, stream>>>(h, aggr,
                                            mw1 + (size_t)l * HID * HID, mb1 + (size_t)l * HID,
                                            mw2 + (size_t)l * HID * HID, mb2 + (size_t)l * HID,
                                            gam + (size_t)l * HID, bet + (size_t)l * HID,
                                            N_NODES);
  }

  k_zero<<<1024, 256, 0, stream>>>(psum, (long)N_GRAPHS * HID);
  k_zero<<<1024, 256, 0, stream>>>((float*)maxb, (long)N_GRAPHS * HID); // 0 == ordered -inf
  k_zero<<<8, 256, 0, stream>>>(cnt, (long)N_GRAPHS);
  k_pool<<<((long)N_NODES * 32 + 255) / 256, 256, 0, stream>>>(h, batch, psum, maxb, cnt);
  k_poolfin<<<(N_GRAPHS * HID + 255) / 256, 256, 0, stream>>>(psum, maxb, cnt, emb, comb);
  k_exp<<<N_GRAPHS, 128, 0, stream>>>(expf, ew1, eb1, ew2, eb2, comb);
  k_head<<<N_GRAPHS, 256, 0, stream>>>(comb, hw1, hb1, hw2, hb2, hw3, hb3, outv);
}
